// GCN_30666066494224
// MI455X (gfx1250) — matrix operand order repaired and checked by reading, not compiled, not run on hardware
//
#include <hip/hip_runtime.h>
#include <hip/hip_bf16.h>

typedef __attribute__((ext_vector_type(16))) __bf16       v16bf;
typedef __attribute__((ext_vector_type(8)))  float        v8f;
typedef __attribute__((ext_vector_type(8)))  unsigned int v8u;

#define N_NODES 50000
#define HIDDEN  256
#define IN_DIM  1000
#define KPAD    1024
#define OUT_DIM 2

__device__ __forceinline__ unsigned short f32_to_bf16(float f) {
    unsigned int u = __float_as_uint(f);
    unsigned int r = u + 0x7FFFu + ((u >> 16) & 1u);   // round-to-nearest-even
    return (unsigned short)(r >> 16);
}

// ---------------------------------------------------------------- utilities
__global__ void gcn_fill_f32(float* __restrict__ p, float v, long n) {
    long i = (long)blockIdx.x * blockDim.x + threadIdx.x;
    if (i < n) p[i] = v;
}

__global__ void gcn_count_deg(const int* __restrict__ dst, float* __restrict__ deg, int E) {
    int e = blockIdx.x * blockDim.x + threadIdx.x;
    if (e < E) atomicAdd(&deg[dst[e]], 1.0f);
}

__global__ void gcn_dinv(float* __restrict__ deg, int n) {
    int i = blockIdx.x * blockDim.x + threadIdx.x;
    if (i < n) {
        float d = deg[i];
        deg[i] = (d > 0.0f) ? rsqrtf(d) : 0.0f;   // in place: deg -> deg^{-1/2}
    }
}

// ------------------------------------------------- W1 -> bf16 fragment order
// Layout: block b = kt*16 + ct holds one 32x16 B-tile as 32 lanes x 16 bf16.
// lane = (n&15) + (k_local>=16 ? 16 : 0), elem j = k_local & 15.
__global__ void gcn_prep_w1(const float* __restrict__ W1, unsigned short* __restrict__ w1f) {
    int idx = blockIdx.x * 256 + threadIdx.x;        // 0 .. 1024*256-1
    int b    = idx >> 9;                             // 512 elems per tile
    int rem  = idx & 511;
    int lane = rem >> 4;
    int j    = rem & 15;
    int kt = b >> 4, ct = b & 15;
    int k = kt * 32 + ((lane >> 4) << 4) + j;
    int n = ct * 16 + (lane & 15);
    float v = (k < IN_DIM) ? W1[k * HIDDEN + n] : 0.0f;
    w1f[idx] = f32_to_bf16(v);
}

// ------------------------------------------------------- GEMM1: h1 = x @ W1
// 256 threads = 8 waves; block = 16 rows x 256 cols; wave w -> col tiles 2w,2w+1.
__global__ void __launch_bounds__(256)
gcn_gemm1_wmma(const float* __restrict__ x,
               const unsigned short* __restrict__ w1f,
               float* __restrict__ h1) {
    __shared__ __align__(32) unsigned short aLds[16 * 32];  // A tile, fragment order

    const int  t    = threadIdx.x;
    const int  wave = t >> 5;
    const int  lane = t & 31;
    const long rowBase = (long)blockIdx.x * 16;
    const int  ct0 = wave * 2;

    v8f acc0 = {};
    v8f acc1 = {};

    for (int kt = 0; kt < KPAD / 32; ++kt) {
        const int k0 = kt * 32;
        // cooperative A load: 16 rows x 32 k, fp32 -> bf16, store in fragment order
        #pragma unroll
        for (int i = 0; i < 2; ++i) {
            int idx = t + i * 256;
            int r   = idx >> 5;       // 0..15
            int kk  = idx & 31;       // 0..31
            int k   = k0 + kk;
            float v = (k < IN_DIM) ? x[(rowBase + r) * IN_DIM + k] : 0.0f;
            int fl = r + ((kk >> 4) & 1) * 16;              // fragment lane
            int fj = kk & 15;                               // fragment element
            aLds[fl * 16 + fj] = f32_to_bf16(v);
        }
        __syncthreads();

        v16bf a  = __builtin_bit_cast(v16bf, *(const v8u*)&aLds[lane * 16]);
        const unsigned short* bp0 = w1f + (((kt * 16 + ct0    ) * 32 + lane) << 4);
        const unsigned short* bp1 = w1f + (((kt * 16 + ct0 + 1) * 32 + lane) << 4);
        v16bf b0 = __builtin_bit_cast(v16bf, *(const v8u*)bp0);
        v16bf b1 = __builtin_bit_cast(v16bf, *(const v8u*)bp1);

        acc0 = __builtin_amdgcn_wmma_f32_16x16x32_bf16(false, a, false, b0,
                                                       (short)0, acc0, false, false);
        acc1 = __builtin_amdgcn_wmma_f32_16x16x32_bf16(false, a, false, b1,
                                                       (short)0, acc1, false, false);
        __syncthreads();
    }

    // C/D layout: VGPR v -> rows {v, v+8}, lanes 0-15 / 16-31, col = lane&15
    const int col    = ct0 * 16 + (lane & 15);
    const int rowOff = (lane >> 4) * 8;
    #pragma unroll
    for (int v = 0; v < 8; ++v) {
        long r = rowBase + rowOff + v;
        h1[r * HIDDEN + col]      = acc0[v];
        h1[r * HIDDEN + col + 16] = acc1[v];
    }
}

// ------------------------------------------------ layer-1 edge aggregation
__global__ void gcn_aggregate1(const int* __restrict__ src, const int* __restrict__ dst,
                               const float* __restrict__ dinv, const float* __restrict__ h1,
                               float* __restrict__ agg1) {
    int e = blockIdx.x;
    int s = src[e], d = dst[e];
    float w = dinv[s] * dinv[d];
    int f = threadIdx.x;
    atomicAdd(&agg1[(long)d * HIDDEN + f], h1[(long)s * HIDDEN + f] * w);
}

// self-loop + bias + relu (in place on agg1)
__global__ void gcn_finish1(float* __restrict__ agg1, const float* __restrict__ h1,
                            const float* __restrict__ dinv, const float* __restrict__ b1) {
    long idx = (long)blockIdx.x * 256 + threadIdx.x;
    if (idx >= (long)N_NODES * HIDDEN) return;
    int i = (int)(idx >> 8);
    int f = (int)(idx & 255);
    float di = dinv[i];
    float v = agg1[idx] + di * di * h1[idx] + b1[f];
    agg1[idx] = fmaxf(v, 0.0f);
}

// ------------------------------------------------ GEMM2: h2 = agg1 @ W2 (256->2)
__global__ void gcn_gemm2(const float* __restrict__ a, const float* __restrict__ W2,
                          float* __restrict__ h2) {
    int gid  = blockIdx.x * blockDim.x + threadIdx.x;
    int node = gid >> 5;
    int lane = gid & 31;
    if (node >= N_NODES) return;
    float a0 = 0.0f, a1 = 0.0f;
    #pragma unroll
    for (int i = 0; i < HIDDEN / 32; ++i) {
        int k = lane + i * 32;
        float v = a[(long)node * HIDDEN + k];
        a0 += v * W2[k * OUT_DIM + 0];
        a1 += v * W2[k * OUT_DIM + 1];
    }
    #pragma unroll
    for (int off = 16; off > 0; off >>= 1) {
        a0 += __shfl_down(a0, off, 32);
        a1 += __shfl_down(a1, off, 32);
    }
    if (lane == 0) {
        h2[node * OUT_DIM + 0] = a0;
        h2[node * OUT_DIM + 1] = a1;
    }
}

// ------------------------------------------------ layer-2 edge aggregation
__global__ void gcn_aggregate2(const int* __restrict__ src, const int* __restrict__ dst,
                               const float* __restrict__ dinv, const float* __restrict__ h2,
                               float* __restrict__ out, int E) {
    int idx = blockIdx.x * 256 + threadIdx.x;
    int e = idx >> 1;
    if (e >= E) return;
    int f = idx & 1;
    int s = src[e], d = dst[e];
    float w = dinv[s] * dinv[d];
    atomicAdd(&out[(long)d * OUT_DIM + f], h2[(long)s * OUT_DIM + f] * w);
}

__global__ void gcn_finish2(float* __restrict__ out, const float* __restrict__ h2,
                            const float* __restrict__ dinv, const float* __restrict__ b2) {
    int idx = blockIdx.x * 256 + threadIdx.x;
    if (idx >= N_NODES * OUT_DIM) return;
    int i = idx >> 1, f = idx & 1;
    float di = dinv[i];
    out[idx] = out[idx] + di * di * h2[idx] + b2[f];
}

// =========================================================================
extern "C" void kernel_launch(void* const* d_in, const int* in_sizes, int n_in,
                              void* d_out, int out_size, void* d_ws, size_t ws_size,
                              hipStream_t stream) {
    const float* x  = (const float*)d_in[0];
    const float* W1 = (const float*)d_in[1];
    const float* b1 = (const float*)d_in[2];
    const float* W2 = (const float*)d_in[3];
    const float* b2 = (const float*)d_in[4];
    const int*   ei = (const int*)d_in[5];
    const int E = in_sizes[5] / 2;
    const int* src = ei;
    const int* dst = ei + E;
    float* out = (float*)d_out;

    // workspace carving (256B aligned)
    char* ws = (char*)d_ws;
    size_t off = 0;
    auto carve = [&](size_t bytes) -> char* {
        char* p = ws + off;
        off = (off + bytes + 255) & ~(size_t)255;
        return p;
    };
    float*          deg  = (float*)carve((size_t)N_NODES * 4);                 // -> dinv
    float*          h1   = (float*)carve((size_t)N_NODES * HIDDEN * 4);
    float*          agg1 = (float*)carve((size_t)N_NODES * HIDDEN * 4);
    float*          h2   = (float*)carve((size_t)N_NODES * OUT_DIM * 4);
    unsigned short* w1f  = (unsigned short*)carve((size_t)KPAD * HIDDEN * 2);
    (void)ws_size;

    // re-initialize accumulation buffers every call (deterministic replays)
    gcn_fill_f32<<<(N_NODES + 255) / 256, 256, 0, stream>>>(deg, 1.0f, N_NODES); // self-loop
    {
        long n = (long)N_NODES * HIDDEN;
        gcn_fill_f32<<<(int)((n + 255) / 256), 256, 0, stream>>>(agg1, 0.0f, n);
    }
    gcn_fill_f32<<<(N_NODES * OUT_DIM + 255) / 256, 256, 0, stream>>>(out, 0.0f,
                                                                      N_NODES * OUT_DIM);

    // W1 -> bf16, fragment-ordered, K padded to 1024
    gcn_prep_w1<<<(KPAD * HIDDEN) / 256, 256, 0, stream>>>(W1, w1f);

    // degrees -> deg^{-1/2}
    gcn_count_deg<<<(E + 255) / 256, 256, 0, stream>>>(dst, deg, E);
    gcn_dinv<<<(N_NODES + 255) / 256, 256, 0, stream>>>(deg, N_NODES);

    // layer 1
    gcn_gemm1_wmma<<<N_NODES / 16, 256, 0, stream>>>(x, w1f, h1);
    gcn_aggregate1<<<E, HIDDEN, 0, stream>>>(src, dst, deg, h1, agg1);
    {
        long n = (long)N_NODES * HIDDEN;
        gcn_finish1<<<(int)((n + 255) / 256), 256, 0, stream>>>(agg1, h1, deg, b1);
    }

    // layer 2
    gcn_gemm2<<<(N_NODES * 32 + 255) / 256, 256, 0, stream>>>(agg1, W2, h2);
    gcn_aggregate2<<<(E * OUT_DIM + 255) / 256, 256, 0, stream>>>(src, dst, deg, h2, out, E);
    gcn_finish2<<<(N_NODES * OUT_DIM + 255) / 256, 256, 0, stream>>>(out, h2, deg, b2);
}
